// MultiHeadAttention_66692252172637
// MI455X (gfx1250) — matrix operand order repaired and checked by reading, not compiled, not run on hardware
//
#include <hip/hip_runtime.h>
#include <cstdint>

// ---------------------------------------------------------------- types
typedef __bf16 bf16;
typedef __attribute__((ext_vector_type(16))) __bf16 v16bf;
typedef __attribute__((ext_vector_type(8)))  __bf16 v8bf;
typedef __attribute__((ext_vector_type(8)))  float  v8f;

#define D_MODEL 1024
#define NHEAD   16
#define EDIM    64
#define BATCH   4
#define SEQ     2048

// ---------------------------------------------------------------- WMMA
__device__ __forceinline__ v8f wmma_bf16(v16bf a, v16bf b, v8f c) {
    return __builtin_amdgcn_wmma_f32_16x16x32_bf16(
        /*neg_a=*/false, a, /*neg_b=*/false, b,
        /*c_mod=*/(short)0, c, /*reuse_a=*/false, /*reuse_b=*/false);
}

// ---------------------------------------------------------------- async copy
// CDNA5 GLOBAL_LOAD_ASYNC_TO_LDS_B128: per-lane 16B memory -> LDS, ASYNCcnt.
__device__ __forceinline__ void async_copy_b128(const void* gptr, void* lptr) {
    // generic LDS pointer: addr[31:0] is the LDS byte offset (aperture rule)
    uint32_t laddr = (uint32_t)(uintptr_t)lptr;
    asm volatile("global_load_async_to_lds_b128 %0, %1, off"
                 :: "v"(laddr), "v"(gptr) : "memory");
}
__device__ __forceinline__ void async_wait0() {
    asm volatile("s_wait_asynccnt 0x0" ::: "memory");
}

// A fragment (16x32 MxK, bf16). base -> element [0][0], ld = row stride (elems).
// lanes 0-15: M=lane, K = {0..7,16..23}; lanes 16-31: M=lane-16, K = {8..15,24..31}
__device__ __forceinline__ v16bf load_frag_a(const bf16* base, int ld) {
    const int lane = threadIdx.x & 31;
    const int row  = lane & 15;
    const int kb   = (lane >> 4) << 3;           // 0 or 8
    const bf16* p  = base + row * ld + kb;
    v8bf c0 = *(const v8bf*)(p);
    v8bf c1 = *(const v8bf*)(p + 16);
    v16bf r;
#pragma unroll
    for (int i = 0; i < 8; ++i) { r[i] = c0[i]; r[i + 8] = c1[i]; }
    return r;
}

// B fragment (32x16 KxN, bf16) from an N-major array: base -> element [n=0][k=0],
// ld = stride between consecutive n (elems). lane%16 = n; lanes0-15 K={0..7,16..23},
// lanes16-31 K={8..15,24..31} (two b128 reads).
__device__ __forceinline__ v16bf load_frag_b(const bf16* base, int ld) {
    const int lane = threadIdx.x & 31;
    const int n    = lane & 15;
    const int kb   = (lane >> 4) << 3;           // 0 or 8
    const bf16* p  = base + n * ld + kb;
    v8bf c0 = *(const v8bf*)(p);
    v8bf c1 = *(const v8bf*)(p + 16);
    v16bf r;
#pragma unroll
    for (int i = 0; i < 8; ++i) { r[i] = c0[i]; r[i + 8] = c1[i]; }
    return r;
}

// ---------------------------------------------------------------- convert
__global__ __launch_bounds__(256) void f32_to_bf16_kernel(
    const float* __restrict__ in, bf16* __restrict__ out, int n) {
    int i = blockIdx.x * 256 + threadIdx.x;
    if (i < n) out[i] = (bf16)in[i];
}

// ---------------------------------------------------------------- GEMM
// C[m,n] = sum_k A[m,k] * Bm[n,k] + bias[n]
// MODE 0: scatter into K/Q/V bf16 [b,h,s,e] buffers (QKV projection, N=3072)
// MODE 1: Cout[m,n] = float (output projection)
#define BM 128
#define BN 128
#define BK 32
#define LDS_STRIDE 48   // 96 bytes per row -> 16B aligned, conflict-padded

template <int MODE>
__global__ __launch_bounds__(256) void gemm_bf16_kernel(
    const bf16* __restrict__ A,    // [M,K]
    const bf16* __restrict__ Bm,   // [N,K]
    const float* __restrict__ bias,// [N]
    float* __restrict__ Cout,      // MODE 1
    bf16* __restrict__ Kb, bf16* __restrict__ Qb, bf16* __restrict__ Vb, // MODE 0
    int M, int N, int K)
{
    __shared__ bf16 As[BM * LDS_STRIDE];
    __shared__ bf16 Bs[BN * LDS_STRIDE];

    const int tid  = threadIdx.x;
    const int wave = tid >> 5;
    const int wm   = wave & 3;       // 4 waves along M (32 rows each)
    const int wn   = wave >> 2;      // 2 waves along N (64 cols each)
    const int m0   = blockIdx.y * BM;
    const int n0   = blockIdx.x * BN;

    v8f acc[2][4];
#pragma unroll
    for (int a = 0; a < 2; ++a)
#pragma unroll
        for (int b = 0; b < 4; ++b)
#pragma unroll
            for (int r = 0; r < 8; ++r) acc[a][b][r] = 0.0f;

    const int srow = tid >> 1;             // 0..127
    const int scol = (tid & 1) * 16;       // 0 or 16

    for (int k0 = 0; k0 < K; k0 += BK) {
        const bf16* ga = A  + (size_t)(m0 + srow) * K + k0 + scol;
        const bf16* gb = Bm + (size_t)(n0 + srow) * K + k0 + scol;
        // async global -> LDS staging (no VGPR round-trip)
        async_copy_b128(ga,     &As[srow * LDS_STRIDE + scol]);
        async_copy_b128(ga + 8, &As[srow * LDS_STRIDE + scol + 8]);
        async_copy_b128(gb,     &Bs[srow * LDS_STRIDE + scol]);
        async_copy_b128(gb + 8, &Bs[srow * LDS_STRIDE + scol + 8]);
        if (k0 + BK < K) {                 // CDNA5 global_prefetch_b8
            __builtin_prefetch(ga + BK, 0, 0);
            __builtin_prefetch(gb + BK, 0, 0);
        }
        async_wait0();
        __syncthreads();

        v16bf a0 = load_frag_a(&As[(wm * 32 + 0)  * LDS_STRIDE], LDS_STRIDE);
        v16bf a1 = load_frag_a(&As[(wm * 32 + 16) * LDS_STRIDE], LDS_STRIDE);
#pragma unroll
        for (int t = 0; t < 4; ++t) {
            v16bf bt = load_frag_b(&Bs[(wn * 64 + t * 16) * LDS_STRIDE], LDS_STRIDE);
            acc[0][t] = wmma_bf16(a0, bt, acc[0][t]);
            acc[1][t] = wmma_bf16(a1, bt, acc[1][t]);
        }
        __syncthreads();
    }

    // epilogue: C layout -> VGPR r holds row (r or r+8), col = lane%16
    const int lane = tid & 31;
    const int ln   = lane & 15;
    const int rhi  = (lane >> 4) * 8;
#pragma unroll
    for (int mt = 0; mt < 2; ++mt) {
#pragma unroll
        for (int nt = 0; nt < 4; ++nt) {
            const int gn = n0 + wn * 64 + nt * 16 + ln;
            const float bv = bias[gn];
#pragma unroll
            for (int r = 0; r < 8; ++r) {
                const int gm = m0 + wm * 32 + mt * 16 + r + rhi;
                const float v = acc[mt][nt][r] + bv;
                if (MODE == 1) {
                    Cout[(size_t)gm * N + gn] = v;
                } else {
                    // source chunk order: (k, q, v)
                    const int chunk = gn >> 10;        // /1024
                    const int d     = gn & 1023;
                    const int head  = d >> 6;
                    const int e     = d & 63;
                    const int batch = gm >> 11;        // /2048
                    const int s     = gm & 2047;
                    bf16* dst = (chunk == 0) ? Kb : (chunk == 1) ? Qb : Vb;
                    dst[((size_t)((batch << 4) + head) * SEQ + s) * EDIM + e] = (bf16)v;
                }
            }
        }
    }
}

// ---------------------------------------------------------------- attention
// Flash-style. Grid: (SEQ/128, BATCH*NHEAD). 8 waves; wave owns 16 query rows.
#define LDK 72   // K-tile LDS row stride (144B, 16B aligned)
#define LDV 48   // transposed-V / P staging row stride (96B, 16B aligned)

__global__ __launch_bounds__(256) void attn_kernel(
    const bf16* __restrict__ Qb, const bf16* __restrict__ Kb,
    const bf16* __restrict__ Vb, const unsigned char* __restrict__ mask,
    bf16* __restrict__ Yb)
{
    __shared__ bf16 Kt[32 * LDK];          // [key][e] row-major
    __shared__ bf16 VtT[64 * LDV];         // [e][key] (transposed)
    __shared__ bf16 Ps[8][16 * LDV];       // per-wave P (16 x 32) staging

    const int tid  = threadIdx.x;
    const int wave = tid >> 5;
    const int lane = tid & 31;
    const int bh   = blockIdx.y;
    const int batch = bh >> 4;
    const int head  = bh & 15;
    const int q0   = blockIdx.x * 128 + wave * 16;

    const bf16* qptr = Qb + ((size_t)bh * SEQ + q0) * EDIM;
    const bf16* kptr = Kb + (size_t)bh * SEQ * EDIM;
    const bf16* vptr = Vb + (size_t)bh * SEQ * EDIM;
    const unsigned char* mrow = mask + batch * SEQ;

    // Q fragments stay in registers for the whole key loop
    const v16bf aq0 = load_frag_a(qptr, EDIM);        // e 0..31
    const v16bf aq1 = load_frag_a(qptr + 32, EDIM);   // e 32..63

    float rmax[8], rsum[8];
#pragma unroll
    for (int r = 0; r < 8; ++r) { rmax[r] = -3.0e38f; rsum[r] = 0.0f; }
    v8f o[4];
#pragma unroll
    for (int t = 0; t < 4; ++t)
#pragma unroll
        for (int r = 0; r < 8; ++r) o[t][r] = 0.0f;

    const int srow = tid >> 3;            // 0..31 (key row)
    const int sc   = (tid & 7) * 8;       // e chunk base

    for (int kb0 = 0; kb0 < SEQ; kb0 += 32) {
        // stage K tile row-major via async global->LDS copy
        async_copy_b128(&kptr[(size_t)(kb0 + srow) * EDIM + sc],
                        &Kt[srow * LDK + sc]);
        // stage V tile transposed: VtT[e][key] (manual scatter)
        v8bf vv = *(const v8bf*)&vptr[(size_t)(kb0 + srow) * EDIM + sc];
#pragma unroll
        for (int i = 0; i < 8; ++i) VtT[(sc + i) * LDV + srow] = vv[i];
        async_wait0();
        __syncthreads();

        // scores: two 16-key tiles, e reduced in two K=32 chunks
        v8f s[2];
#pragma unroll
        for (int nt = 0; nt < 2; ++nt) {
            v8f t;
#pragma unroll
            for (int r = 0; r < 8; ++r) t[r] = 0.0f;
            t = wmma_bf16(aq0, load_frag_b(&Kt[(nt * 16) * LDK + 0],  LDK), t);
            t = wmma_bf16(aq1, load_frag_b(&Kt[(nt * 16) * LDK + 32], LDK), t);
            s[nt] = t;
        }

        const float mb0 = mrow[kb0 + (lane & 15)]      ? 0.0f : -1.0e9f;
        const float mb1 = mrow[kb0 + 16 + (lane & 15)] ? 0.0f : -1.0e9f;
        const int prow  = ((lane >> 4) ? 8 : 0);

        // online softmax: row r lives across 16 lanes in vector element r
#pragma unroll
        for (int r = 0; r < 8; ++r) {
            float s0 = s[0][r] * 0.125f + mb0;   // 1/sqrt(64)
            float s1 = s[1][r] * 0.125f + mb1;
            float rm = fmaxf(s0, s1);
            rm = fmaxf(rm, __shfl_xor(rm, 1, 32));
            rm = fmaxf(rm, __shfl_xor(rm, 2, 32));
            rm = fmaxf(rm, __shfl_xor(rm, 4, 32));
            rm = fmaxf(rm, __shfl_xor(rm, 8, 32));
            const float mnew = fmaxf(rmax[r], rm);
            const float corr = __expf(rmax[r] - mnew);
            const float p0 = __expf(s0 - mnew);
            const float p1 = __expf(s1 - mnew);
            float rs = p0 + p1;
            rs += __shfl_xor(rs, 1, 32);
            rs += __shfl_xor(rs, 2, 32);
            rs += __shfl_xor(rs, 4, 32);
            rs += __shfl_xor(rs, 8, 32);
            rsum[r] = rsum[r] * corr + rs;
            rmax[r] = mnew;
#pragma unroll
            for (int t = 0; t < 4; ++t) o[t][r] *= corr;
            // C-layout -> A-layout via per-wave LDS staging (bf16)
            Ps[wave][(r + prow) * LDV + (lane & 15)]      = (bf16)p0;
            Ps[wave][(r + prow) * LDV + 16 + (lane & 15)] = (bf16)p1;
        }

        // PV: A = P (16x32), B = V^T tiles (32 keys x 16 e)
        const v16bf ap = load_frag_a(&Ps[wave][0], LDV);
#pragma unroll
        for (int t = 0; t < 4; ++t)
            o[t] = wmma_bf16(ap, load_frag_b(&VtT[(t * 16) * LDV], LDV), o[t]);

        __syncthreads();
    }

    // normalize and write y in [b, s, h*e] layout (bf16 for final GEMM)
#pragma unroll
    for (int r = 0; r < 8; ++r) {
        const float inv = 1.0f / rsum[r];
        const int gs = q0 + r + ((lane >> 4) ? 8 : 0);
        const size_t rowbase = ((size_t)batch * SEQ + gs) * D_MODEL + head * EDIM;
#pragma unroll
        for (int t = 0; t < 4; ++t)
            Yb[rowbase + t * 16 + (lane & 15)] = (bf16)(o[t][r] * inv);
    }
}

// ---------------------------------------------------------------- launch
extern "C" void kernel_launch(void* const* d_in, const int* in_sizes, int n_in,
                              void* d_out, int out_size, void* d_ws, size_t ws_size,
                              hipStream_t stream) {
    const float* x      = (const float*)d_in[0];
    const unsigned char* mask = (const unsigned char*)d_in[1];
    const float* W_qkv  = (const float*)d_in[2];
    const float* b_qkv  = (const float*)d_in[3];
    const float* W_proj = (const float*)d_in[4];
    const float* b_proj = (const float*)d_in[5];
    float* out = (float*)d_out;

    const int M = BATCH * SEQ;          // 8192
    const int NQ = 3 * D_MODEL;         // 3072
    const int K = D_MODEL;              // 1024

    // workspace carve (256B aligned)
    size_t off = 0;
    auto carve = [&](size_t bytes) {
        void* p = (char*)d_ws + off;
        off += (bytes + 255) & ~(size_t)255;
        return p;
    };
    bf16* xb  = (bf16*)carve((size_t)M * K * 2);
    bf16* Wqb = (bf16*)carve((size_t)NQ * K * 2);
    bf16* Wpb = (bf16*)carve((size_t)D_MODEL * K * 2);
    bf16* Qb  = (bf16*)carve((size_t)BATCH * NHEAD * SEQ * EDIM * 2);
    bf16* Kb  = (bf16*)carve((size_t)BATCH * NHEAD * SEQ * EDIM * 2);
    bf16* Vb  = (bf16*)carve((size_t)BATCH * NHEAD * SEQ * EDIM * 2);
    bf16* Yb  = (bf16*)carve((size_t)M * D_MODEL * 2);
    (void)ws_size;

    // 1) fp32 -> bf16 conversions
    {
        int n;
        n = M * K;            f32_to_bf16_kernel<<<(n + 255) / 256, 256, 0, stream>>>(x, xb, n);
        n = NQ * K;           f32_to_bf16_kernel<<<(n + 255) / 256, 256, 0, stream>>>(W_qkv, Wqb, n);
        n = D_MODEL * K;      f32_to_bf16_kernel<<<(n + 255) / 256, 256, 0, stream>>>(W_proj, Wpb, n);
    }

    // 2) fused QKV projection -> K/Q/V bf16 [b,h,s,e]
    gemm_bf16_kernel<0><<<dim3(NQ / BN, M / BM), 256, 0, stream>>>(
        xb, Wqb, b_qkv, nullptr, Kb, Qb, Vb, M, NQ, K);

    // 3) flash attention -> Yb bf16 [b,s,d]
    attn_kernel<<<dim3(SEQ / 128, BATCH * NHEAD), 256, 0, stream>>>(
        Qb, Kb, Vb, mask, Yb);

    // 4) output projection -> fp32 d_out
    gemm_bf16_kernel<1><<<dim3(D_MODEL / BN, M / BM), 256, 0, stream>>>(
        Yb, Wpb, b_proj, out, nullptr, nullptr, nullptr, M, D_MODEL, K);
}
